// E_GCL_AT_11751030522786
// MI455X (gfx1250) — compile-verified, matrix operand order checked by reading, not run on hardware
//
#include <hip/hip_runtime.h>
#include <hip/hip_bf16.h>
#include <math.h>

typedef _Float16 half_t;
typedef __attribute__((ext_vector_type(16))) _Float16 v16h;
typedef __attribute__((ext_vector_type(8)))  _Float16 v8h;
typedef __attribute__((ext_vector_type(8)))  float    v8f;
typedef __attribute__((ext_vector_type(4)))  unsigned int v4u;
typedef __attribute__((ext_vector_type(8)))  int      v8i;
typedef __attribute__((ext_vector_type(4)))  int      v4i;

#define BATCH 4096
#define SEQ   64
#define DIM   256
#define BT    (BATCH * SEQ)          // 262144 rows

// TDM (Tensor Data Mover): this toolchain exposes the 6-arg builtin
// (uint32x4 g0, int32x8 g1, int32x4 g2, int32x4 g3, int32x8 g4, i32 cpol).
#if defined(__HIP_DEVICE_COMPILE__) && __has_builtin(__builtin_amdgcn_tensor_load_to_lds) && __has_builtin(__builtin_amdgcn_s_wait_tensorcnt)
#define USE_TDM 1
#else
#define USE_TDM 0
#endif

// ---------------------------------------------------------------------------
// Fragment loader for 16-bit A/B operands of V_WMMA_F32_16X16X32_F16.
// Per ISA 7.12.2 (16-bit A 16x32): lane L (0-15) holds row M=L with
// K = {c..c+7} in VGPR0-3 and K = {c+16..c+23} in VGPR4-7 where c = kbase;
// lanes 16-31 hold the same rows with c = kbase+8. `p` must point at
// row*ld + kbase + (lane>=16 ? 8 : 0). B fragments are identical loads from
// rows of B^T (N x K, row-major). Works for global or LDS pointers.
// ---------------------------------------------------------------------------
__device__ __forceinline__ v16h load_frag(const half_t* p) {
    v8h lo = *(const v8h*)(p);        // K = c .. c+7
    v8h hi = *(const v8h*)(p + 16);   // K = c+16 .. c+23
    v16h r;
#pragma unroll
    for (int i = 0; i < 8; ++i) { r[i] = lo[i]; r[8 + i] = hi[i]; }
    return r;
}

__device__ __forceinline__ v8f wmma_f16(v16h a, v16h b, v8f c) {
    return __builtin_amdgcn_wmma_f32_16x16x32_f16(false, a, false, b,
                                                  (short)0, c, false, false);
}

__device__ __forceinline__ v8f vzero8() {
    v8f z = {0.f, 0.f, 0.f, 0.f, 0.f, 0.f, 0.f, 0.f};
    return z;
}

// ---------------------------------------------------------------------------
// Kernel: f32 -> f16 cast of h
// ---------------------------------------------------------------------------
__global__ __launch_bounds__(256)
void cvt_h_kernel(const float* __restrict__ src, half_t* __restrict__ dst, int n) {
    int i = blockIdx.x * 256 + threadIdx.x;
    if (i < n) dst[i] = (half_t)src[i];
}

// ---------------------------------------------------------------------------
// Kernel: transpose+cast the 7 DxD weight matrices into f16 N-major (N x K)
// so WMMA B fragments load exactly like A fragments.
// wt[m*65536 + n*256 + k] = W_m[k*256 + n]
// ---------------------------------------------------------------------------
__global__ __launch_bounds__(256)
void cvt_w_kernel(const float* __restrict__ W0, const float* __restrict__ W1,
                  const float* __restrict__ W2, const float* __restrict__ W3,
                  const float* __restrict__ W4, const float* __restrict__ W5,
                  const float* __restrict__ W6, half_t* __restrict__ wt) {
    int id = blockIdx.x * 256 + threadIdx.x;
    if (id >= 7 * 65536) return;
    int m = id >> 16;
    int r = id & 65535;
    int n = r >> 8;
    int k = r & 255;
    const float* W;
    switch (m) {
        case 0: W = W0; break; case 1: W = W1; break; case 2: W = W2; break;
        case 3: W = W3; break; case 4: W = W4; break; case 5: W = W5; break;
        default: W = W6; break;
    }
    wt[id] = (half_t)W[k * 256 + n];
}

// ---------------------------------------------------------------------------
// Kernel: Y = relu(X @ W + b).  X: (nrows x 256) f16, Wt: W^T (256 x 256) f16
// staged whole in LDS (128 KB) via the Tensor Data Mover (one TDM op from
// wave 0, S_WAIT_TENSORCNT), X tile (64 x 256, 32 KB) staged cooperatively.
// 8 waves; each wave owns a 16x128 strip (8 accumulator tiles); K-loop of 8
// -> 64 WMMA / wave, all operand traffic from LDS.
// ---------------------------------------------------------------------------
__global__ __launch_bounds__(256)
void gemm_relu_256(const half_t* __restrict__ X, const half_t* __restrict__ Wt,
                   const float* __restrict__ bias, half_t* __restrict__ Y) {
    extern __shared__ char smem[];
    half_t* sW = (half_t*)smem;                 // 256*256 f16 = 128 KB
    half_t* sX = (half_t*)(smem + 131072);      // 64*256  f16 =  32 KB

    const int tid  = threadIdx.x;
    const int wave = tid >> 5;
    const int lane = tid & 31;
    const int l15  = lane & 15;
    const int koff = (lane >= 16) ? 8 : 0;   // K sub-offset AND C-row offset

    // ---- stage X tile (all waves) ----
    {
        const uint4* gx = (const uint4*)(X + (size_t)blockIdx.x * 64 * 256);
        uint4* lx = (uint4*)sX;
        for (int i = tid; i < 2048; i += 256) lx[i] = gx[i];
    }

    // ---- stage full weight matrix into LDS ----
#if USE_TDM
    if (tid < 32) {
        // Tensor DMA descriptor (D#), 2D tile: 256 x 256 f16, contiguous.
        unsigned lds = (unsigned)(size_t)sW;
        unsigned long long ga = (unsigned long long)(size_t)Wt;
        v4u g0 = { 1u,                                   // count=1, user mode
                   lds,                                  // lds_addr [63:32]
                   (unsigned)(ga & 0xFFFFFFFFu),         // global_addr lo
                   (unsigned)((ga >> 32) & 0x1FFFFFFu) | (2u << 30) }; // hi | type=2
        v8i g1 = { (int)0x00010000,        // workgroup_mask=0, data_size=1 (2B)
                   (int)(256u << 16),      // tensor_dim0[15:0] @ bits[63:48]
                   (int)(256u << 16),      // tensor_dim1[15:0] @ bits[95:80]
                   (int)(256u << 16),      // tile_dim0=256     @ bits[127:112]
                   (int)256,               // tile_dim1=256     @ bits[143:128]
                   (int)256,               // tensor_dim0_stride=256 (low 32)
                   0, 0 };
        v4i g2 = {0, 0, 0, 0};
        v4i g3 = {0, 0, 0, 0};
        v8i g4 = {0, 0, 0, 0, 0, 0, 0, 0};
        __builtin_amdgcn_tensor_load_to_lds(g0, g1, g2, g3, g4, 0);
        __builtin_amdgcn_s_wait_tensorcnt(0);
    }
#else
    {
        const uint4* gw = (const uint4*)Wt;
        uint4* lw = (uint4*)sW;
        for (int i = tid; i < 8192; i += 256) lw[i] = gw[i];
    }
#endif
    __syncthreads();

    const int mt = wave & 3;       // which 16-row tile within the 64-row block
    const int nh = wave >> 2;      // which 128-col half

    v8f acc[8];
#pragma unroll
    for (int i = 0; i < 8; ++i) acc[i] = vzero8();

    const half_t* xrow = sX + (mt * 16 + l15) * 256;

#pragma unroll
    for (int kt = 0; kt < 8; ++kt) {
        v16h a = load_frag(xrow + kt * 32 + koff);
#pragma unroll
        for (int nt = 0; nt < 8; ++nt) {
            int ncol = nh * 128 + nt * 16 + l15;
            v16h b = load_frag(sW + ncol * 256 + kt * 32 + koff);
            acc[nt] = wmma_f16(a, b, acc[nt]);
        }
    }

    // Epilogue: C layout -> VGPR r is (M = r + koff, N = l15) for this lane.
#pragma unroll
    for (int nt = 0; nt < 8; ++nt) {
        int ncol = nh * 128 + nt * 16 + l15;
        float bv = bias[ncol];
#pragma unroll
        for (int r = 0; r < 8; ++r) {
            int orow = blockIdx.x * 64 + mt * 16 + r + koff;
            float v = acc[nt][r] + bv;
            v = v > 0.f ? v : 0.f;
            Y[(size_t)orow * 256 + ncol] = (half_t)v;
        }
    }
}

// ---------------------------------------------------------------------------
// Kernel: cv[row] = sum_d cc[row,d] * wc2[d]   (wave-per-row, wave32 shuffle)
// ---------------------------------------------------------------------------
__global__ __launch_bounds__(256)
void coordvec_kernel(const half_t* __restrict__ cc, const float* __restrict__ wc2,
                     float* __restrict__ cv) {
    int row  = blockIdx.x * 8 + (threadIdx.x >> 5);
    int lane = threadIdx.x & 31;
    float s = 0.f;
#pragma unroll
    for (int i = 0; i < 8; ++i) {
        int d = lane + i * 32;
        s += (float)cc[(size_t)row * 256 + d] * wc2[d];
    }
#pragma unroll
    for (int m = 16; m >= 1; m >>= 1) s += __shfl_xor(s, m, 32);
    if (lane == 0) cv[row] = s;
}

// ---------------------------------------------------------------------------
// Fused attention: one workgroup per batch element b.
//   S = q k^T  (WMMA, 64x64, K=256)  -> softmax (f32, LDS)
//   h_out = h + alpha @ v            (WMMA, 64x256, K=64)
//   coord_out = coord + sum_j alpha*cv_j*(coord_j - coord_i)
// Dynamic LDS layout (153 KB): sQ|sK|sV|sVT (f16) + sS (f32) + sA (f16) + misc
// ---------------------------------------------------------------------------
__global__ __launch_bounds__(256)
void attention_kernel(const half_t* __restrict__ q16, const half_t* __restrict__ k16,
                      const half_t* __restrict__ v16, const float* __restrict__ cvg,
                      const float* __restrict__ h_in, const float* __restrict__ coord_in,
                      float* __restrict__ out_h, float* __restrict__ out_c) {
    extern __shared__ char smem[];
    half_t* sQ  = (half_t*)smem;              // 64*256
    half_t* sK  = sQ  + 64 * 256;
    half_t* sV  = sK  + 64 * 256;
    half_t* sVT = sV  + 64 * 256;             // 256*64 (v transposed)
    float*  sS  = (float*)(sVT + 256 * 64);   // 64*64 scores -> alpha (f32)
    half_t* sA  = (half_t*)(sS + 64 * 64);    // 64*64 alpha (f16)
    float*  sC  = (float*)(sA + 64 * 64);     // 64*3 coords
    float*  scv = sC + 64 * 3;                // 64

    const int b    = blockIdx.x;
    const int tid  = threadIdx.x;
    const int wave = tid >> 5;
    const int lane = tid & 31;
    const int l15  = lane & 15;
    const int koff = (lane >= 16) ? 8 : 0;

    // ---- stage q,k,v (16B vector copies), cv, coord ----
    {
        const uint4* gq = (const uint4*)(q16 + (size_t)b * 64 * 256);
        const uint4* gk = (const uint4*)(k16 + (size_t)b * 64 * 256);
        const uint4* gv = (const uint4*)(v16 + (size_t)b * 64 * 256);
        uint4* lq = (uint4*)sQ; uint4* lk = (uint4*)sK; uint4* lv = (uint4*)sV;
        for (int i = tid; i < 2048; i += 256) { lq[i] = gq[i]; lk[i] = gk[i]; lv[i] = gv[i]; }
        if (tid < 64)  scv[tid] = cvg[b * 64 + tid];
        if (tid < 192) sC[tid]  = coord_in[(size_t)b * 192 + tid];
    }
    __syncthreads();

    // ---- transpose V in LDS: sVT[d*64 + j] = sV[j*256 + d] ----
    for (int i = tid; i < 64 * 256; i += 256) {
        int j = i >> 8, d = i & 255;
        sVT[d * 64 + j] = sV[j * 256 + d];
    }
    __syncthreads();

    // ---- S = q k^T : 4x4 tiles of 16x16, each wave does 2 tiles ----
    {
        const int mt  = wave >> 1;
        const int nt0 = (wave & 1) * 2;
        v8f c0 = vzero8(), c1 = vzero8();
        const half_t* qrow  = sQ + (mt * 16 + l15) * 256;
        const half_t* k0row = sK + ((nt0 + 0) * 16 + l15) * 256;  // B^T rows = k rows
        const half_t* k1row = sK + ((nt0 + 1) * 16 + l15) * 256;
#pragma unroll
        for (int kt = 0; kt < 8; ++kt) {
            v16h a  = load_frag(qrow  + kt * 32 + koff);
            v16h b0 = load_frag(k0row + kt * 32 + koff);
            v16h b1 = load_frag(k1row + kt * 32 + koff);
            c0 = wmma_f16(a, b0, c0);
            c1 = wmma_f16(a, b1, c1);
        }
#pragma unroll
        for (int r = 0; r < 8; ++r) {
            int orow = mt * 16 + r + koff;
            sS[orow * 64 + (nt0 + 0) * 16 + l15] = c0[r];
            sS[orow * 64 + (nt0 + 1) * 16 + l15] = c1[r];
        }
    }
    __syncthreads();

    // ---- row softmax (f32) ; keep alpha in sS (f32) and sA (f16) ----
    if (tid < 64) {
        float mx = -3.4e38f;
        for (int j = 0; j < 64; ++j) mx = fmaxf(mx, sS[tid * 64 + j]);
        float sum = 0.f;
        for (int j = 0; j < 64; ++j) { float e = __expf(sS[tid * 64 + j] - mx); sS[tid * 64 + j] = e; sum += e; }
        float inv = 1.f / sum;
        for (int j = 0; j < 64; ++j) {
            float a = sS[tid * 64 + j] * inv;
            sS[tid * 64 + j] = a;
            sA[tid * 64 + j] = (half_t)a;
        }
    }
    __syncthreads();

    // ---- h_agg = alpha @ v : 4x16 tiles, each wave: 16-row tile x 128 cols ----
    {
        const int mt = wave & 3;
        const int nh = wave >> 2;
        v8f acc[8];
#pragma unroll
        for (int i = 0; i < 8; ++i) acc[i] = vzero8();
        const half_t* arow = sA + (mt * 16 + l15) * 64;
#pragma unroll
        for (int kt = 0; kt < 2; ++kt) {
            v16h a = load_frag(arow + kt * 32 + koff);
#pragma unroll
            for (int nt = 0; nt < 8; ++nt) {
                int d = nh * 128 + nt * 16 + l15;
                v16h bf = load_frag(sVT + d * 64 + kt * 32 + koff);  // B^T rows = v columns
                acc[nt] = wmma_f16(a, bf, acc[nt]);
            }
        }
#pragma unroll
        for (int nt = 0; nt < 8; ++nt) {
            int d = nh * 128 + nt * 16 + l15;
#pragma unroll
            for (int r = 0; r < 8; ++r) {
                int irow = mt * 16 + r + koff;
                size_t gi = ((size_t)b * 64 + irow) * 256 + d;
                out_h[gi] = h_in[gi] + acc[nt][r];
            }
        }
    }

    // ---- coord update (uses f32 alpha in sS; no LDS writes since barrier) ----
    if (tid < 64) {
        float cx = sC[tid * 3 + 0], cy = sC[tid * 3 + 1], cz = sC[tid * 3 + 2];
        float ax = 0.f, ay = 0.f, az = 0.f;
        for (int j = 0; j < 64; ++j) {
            float w = sS[tid * 64 + j] * scv[j];
            ax += w * (sC[j * 3 + 0] - cx);
            ay += w * (sC[j * 3 + 1] - cy);
            az += w * (sC[j * 3 + 2] - cz);
        }
        size_t gi = ((size_t)b * 64 + tid) * 3;
        out_c[gi + 0] = cx + ax;
        out_c[gi + 1] = cy + ay;
        out_c[gi + 2] = cz + az;
    }
}

// ---------------------------------------------------------------------------
// Host launcher
// ---------------------------------------------------------------------------
extern "C" void kernel_launch(void* const* d_in, const int* in_sizes, int n_in,
                              void* d_out, int out_size, void* d_ws, size_t ws_size,
                              hipStream_t stream) {
    const float* h     = (const float*)d_in[0];
    const float* coord = (const float*)d_in[1];
    const float* Wq1 = (const float*)d_in[2];  const float* bq1 = (const float*)d_in[3];
    const float* Wq2 = (const float*)d_in[4];  const float* bq2 = (const float*)d_in[5];
    const float* Wk1 = (const float*)d_in[6];  const float* bk1 = (const float*)d_in[7];
    const float* Wk2 = (const float*)d_in[8];  const float* bk2 = (const float*)d_in[9];
    const float* Wv1 = (const float*)d_in[10]; const float* bv1 = (const float*)d_in[11];
    const float* Wv2 = (const float*)d_in[12]; const float* bv2 = (const float*)d_in[13];
    const float* Wc1 = (const float*)d_in[14]; const float* bc1 = (const float*)d_in[15];
    const float* wc2 = (const float*)d_in[16];

    // workspace layout
    half_t* w16 = (half_t*)d_ws;                              // 7 * 65536 f16
    half_t* h16 = (half_t*)((char*)d_ws + (1u << 20));        // BT*256 f16
    half_t* hid = h16 + (size_t)BT * 256;                     // BT*256 f16 (reused)
    half_t* q16 = hid + (size_t)BT * 256;
    half_t* k16 = q16 + (size_t)BT * 256;
    half_t* v16 = k16 + (size_t)BT * 256;
    float*  cv  = (float*)(v16 + (size_t)BT * 256);           // BT f32

    float* out_h = (float*)d_out;                             // BT*256
    float* out_c = out_h + (size_t)BT * 256;                  // BT*3

    const int nElemH = BT * 256;   // 67,108,864
    cvt_h_kernel<<<(nElemH + 255) / 256, 256, 0, stream>>>(h, h16, nElemH);
    cvt_w_kernel<<<(7 * 65536 + 255) / 256, 256, 0, stream>>>(
        Wq1, Wq2, Wk1, Wk2, Wv1, Wv2, Wc1, w16);

    const int gemmGrid = BT / 64;  // 4096
    const size_t gemm_lds = 131072 + 32768;   // sW + sX = 160 KB
    // q = relu(relu(h@Wq1+bq1)@Wq2+bq2)
    gemm_relu_256<<<gemmGrid, 256, gemm_lds, stream>>>(h16, w16 + 0 * 65536, bq1, hid);
    gemm_relu_256<<<gemmGrid, 256, gemm_lds, stream>>>(hid, w16 + 1 * 65536, bq2, q16);
    // k
    gemm_relu_256<<<gemmGrid, 256, gemm_lds, stream>>>(h16, w16 + 2 * 65536, bk1, hid);
    gemm_relu_256<<<gemmGrid, 256, gemm_lds, stream>>>(hid, w16 + 3 * 65536, bk2, k16);
    // v
    gemm_relu_256<<<gemmGrid, 256, gemm_lds, stream>>>(h16, w16 + 4 * 65536, bv1, hid);
    gemm_relu_256<<<gemmGrid, 256, gemm_lds, stream>>>(hid, w16 + 5 * 65536, bv2, v16);
    // cc = relu(v@Wc1+bc1) ; cv = cc @ wc2
    gemm_relu_256<<<gemmGrid, 256, gemm_lds, stream>>>(v16, w16 + 6 * 65536, bc1, hid);
    coordvec_kernel<<<BT / 8, 256, 0, stream>>>(hid, wc2, cv);

    // fused attention + coord update
    const size_t attn_lds =
        (size_t)(3 * 64 * 256 + 256 * 64) * sizeof(half_t)   // sQ,sK,sV,sVT
        + (size_t)(64 * 64) * sizeof(float)                  // sS
        + (size_t)(64 * 64) * sizeof(half_t)                 // sA
        + (size_t)(64 * 3 + 64) * sizeof(float);             // sC, scv
    attention_kernel<<<BATCH, 256, attn_lds, stream>>>(
        q16, k16, v16, cv, h, coord, out_h, out_c);
}
